// CustomNetwork_22651657519163
// MI455X (gfx1250) — matrix operand order repaired and checked by reading, not compiled, not run on hardware
//
#include <hip/hip_runtime.h>
#include <hip/hip_bf16.h>

typedef __attribute__((ext_vector_type(16))) _Float16 v16h;
typedef __attribute__((ext_vector_type(8)))  float    v8f;

#define C_S_CONST 0.3826834323650898f   // sin(pi/8)
#define C_X_CONST 0.9238795325112867f   // cos(pi/8)
#define INV_SQRT3_CONST 0.5773502691896258f
#define RS10 0.31622776601683794f
#define RS16 0.25f
#define RS32 0.17677669529663687f
#define RS48 0.14433756729740643f
#define RS100 0.1f
#define PI_F 3.14159265358979323846f

__device__ __forceinline__ float sigmoidf_(float x) { return 1.f / (1.f + __expf(-x)); }
__device__ __forceinline__ float siluf_(float x) { return x * sigmoidf_(x); }
__device__ __forceinline__ void atomAddF(float* p, float v) {
  (void)__hip_atomic_fetch_add(p, v, __ATOMIC_RELAXED, __HIP_MEMORY_SCOPE_AGENT);
}

// CDNA5 async global->LDS copy (GVS mode: SGPR base + per-lane 32-bit offset),
// tracked by ASYNCcnt.
__device__ __forceinline__ void async_b128(unsigned lds_byte_addr, const void* base,
                                           int byte_off) {
  asm volatile("global_load_async_to_lds_b128 %0, %1, %2 offset:0"
               :: "v"(lds_byte_addr), "v"(byte_off), "s"(base)
               : "memory");
}
__device__ __forceinline__ void wait_async0() {
#if __has_builtin(__builtin_amdgcn_s_wait_asynccnt)
  __builtin_amdgcn_s_wait_asynccnt(0);
#else
  asm volatile("s_wait_asynccnt 0" ::: "memory");
#endif
}

// ---------------------------------------------------------------------------
// Edge geometry: es (E x 4), ev (E x 3), emb (E x 10)
// ---------------------------------------------------------------------------
__global__ __launch_bounds__(256) void edge_geom_kernel(
    const float* __restrict__ pos, const float* __restrict__ eattr,
    const int* __restrict__ ei, float* __restrict__ es, float* __restrict__ ev,
    float* __restrict__ emb, int E) {
  int e = blockIdx.x * 256 + threadIdx.x;
  if (e >= E) return;
  int s = ei[e], d = ei[E + e];
  float dx = pos[(size_t)s * 3 + 0] - pos[(size_t)d * 3 + 0];
  float dy = pos[(size_t)s * 3 + 1] - pos[(size_t)d * 3 + 1];
  float dz = pos[(size_t)s * 3 + 2] - pos[(size_t)d * 3 + 2];
  float r = sqrtf(dx * dx + dy * dy + dz * dz + 1e-12f);
  float inv = 1.f / r;
  float u = 2.f * (r * (1.f / 3.5f) - 1.f);
  float cut = (u > 0.f) ? 0.f
             : ((u < -1.f) ? 1.f : 0.5f * (1.f - __cosf(PI_F * u)));
  es[(size_t)e * 4 + 0] = eattr[(size_t)e * 3 + 0];
  es[(size_t)e * 4 + 1] = eattr[(size_t)e * 3 + 1];
  es[(size_t)e * 4 + 2] = eattr[(size_t)e * 3 + 2];
  es[(size_t)e * 4 + 3] = cut;
  float cv = cut * 1.7320508075688772f * inv;
  ev[(size_t)e * 3 + 0] = cv * dx;
  ev[(size_t)e * 3 + 1] = cv * dy;
  ev[(size_t)e * 3 + 2] = cv * dz;
  const float step = 3.5f / 9.f;
  const float sc = 2.8234621965789103f;  // sqrt(10)/1.12
#pragma unroll
  for (int k = 0; k < 10; k++) {
    float t = (r - (float)k * step) / step;
    emb[(size_t)e * 10 + k] = __expf(-t * t) * sc;
  }
}

// ---------------------------------------------------------------------------
// Node embedding: xs = onehot @ W_embed + b  (N x 16)
// ---------------------------------------------------------------------------
__global__ __launch_bounds__(256) void embed_kernel(
    const float* __restrict__ onehot, const float* __restrict__ W,
    const float* __restrict__ b, float* __restrict__ xs, int N) {
  __shared__ float sW[118 * 16];
  __shared__ float sb[16];
  for (int i = threadIdx.x; i < 118 * 16; i += 256) sW[i] = W[i];
  if (threadIdx.x < 16) sb[threadIdx.x] = b[threadIdx.x];
  __syncthreads();
  int idx = blockIdx.x * 256 + threadIdx.x;
  if (idx >= N * 16) return;
  int n = idx >> 4, j = idx & 15;
  float acc = sb[j];
  for (int k = 0; k < 118; k++) acc += onehot[(size_t)n * 118 + k] * sW[k * 16 + j];
  xs[idx] = acc;
}

// ---------------------------------------------------------------------------
// Per-layer R2 pre-conversion: B16[n][k] (k-major, stride 132, zero-padded
// k in [100,132)), 0.1 = 1/sqrt(100) folded. Done once per layer.
// ---------------------------------------------------------------------------
__global__ __launch_bounds__(256) void convert_R2_kernel(
    const float* __restrict__ R2, _Float16* __restrict__ B16, int WOUT) {
  int idx = blockIdx.x * 256 + threadIdx.x;
  if (idx >= WOUT * 132) return;
  int n = idx / 132, k = idx % 132;
  float v = (k < 100) ? R2[(size_t)k * WOUT + n] * RS100 : 0.f;
  B16[idx] = (_Float16)v;
}

// ---------------------------------------------------------------------------
// Per-layer node pre-pass: sc_scal, (sc_vec), xs1, (xv1), zero ys/yv
// ---------------------------------------------------------------------------
template <int S, int WSC, bool HASV, bool HASSCV>
__global__ __launch_bounds__(128) void node_pre_kernel(
    const float* __restrict__ xs, const float* __restrict__ xv,
    const float* __restrict__ sc_s, const float* __restrict__ sc_v,
    const float* __restrict__ l1s, const float* __restrict__ l1v,
    float* __restrict__ sc_scal, float* __restrict__ sc_vec,
    float* __restrict__ xs1, float* __restrict__ xv1,
    float* __restrict__ ys, float* __restrict__ yv, int N) {
  __shared__ float s_scs[S * WSC];
  __shared__ float s_l1s[S * S];
  __shared__ float s_scv[256];
  __shared__ float s_l1v[256];
  for (int i = threadIdx.x; i < S * WSC; i += 128) s_scs[i] = sc_s[i];
  for (int i = threadIdx.x; i < S * S; i += 128) s_l1s[i] = l1s[i];
  if (HASV)   for (int i = threadIdx.x; i < 256; i += 128) s_l1v[i] = l1v[i];
  if (HASSCV) for (int i = threadIdx.x; i < 256; i += 128) s_scv[i] = sc_v[i];
  __syncthreads();
  int n = blockIdx.x * 128 + threadIdx.x;
  if (n >= N) return;
  const float rsS = (S == 16) ? RS16 : RS32;
  float xr[S];
#pragma unroll
  for (int i = 0; i < S; i++) xr[i] = xs[(size_t)n * S + i];
  for (int j = 0; j < WSC; j++) {
    float acc = 0.f;
#pragma unroll
    for (int i = 0; i < S; i++) acc += xr[i] * s_scs[i * WSC + j];
    sc_scal[(size_t)n * WSC + j] = rsS * acc;
  }
  for (int j = 0; j < S; j++) {
    float acc = 0.f;
#pragma unroll
    for (int i = 0; i < S; i++) acc += xr[i] * s_l1s[i * S + j];
    xs1[(size_t)n * S + j] = rsS * acc;
  }
  if (HASV) {
    float vr[48];
#pragma unroll
    for (int i = 0; i < 48; i++) vr[i] = xv[(size_t)n * 48 + i];
    for (int ww = 0; ww < 16; ww++) {
#pragma unroll
      for (int cc = 0; cc < 3; cc++) {
        float acc = 0.f;
#pragma unroll
        for (int vv = 0; vv < 16; vv++) acc += vr[vv * 3 + cc] * s_l1v[vv * 16 + ww];
        xv1[(size_t)n * 48 + ww * 3 + cc] = 0.25f * acc;
      }
    }
    if (HASSCV) {
      for (int ww = 0; ww < 16; ww++) {
#pragma unroll
        for (int cc = 0; cc < 3; cc++) {
          float acc = 0.f;
#pragma unroll
          for (int vv = 0; vv < 16; vv++) acc += vr[vv * 3 + cc] * s_scv[vv * 16 + ww];
          sc_vec[(size_t)n * 48 + ww * 3 + cc] = 0.25f * acc;
        }
      }
    }
  }
#pragma unroll
  for (int i = 0; i < 48; i++) ys[(size_t)n * 48 + i] = 0.f;
#pragma unroll
  for (int i = 0; i < 144; i++) yv[(size_t)n * 144 + i] = 0.f;
}

// ---------------------------------------------------------------------------
// Edge conv: 8 waves/block, one wave owns 16 edges. Hidden layer in VALU,
// h staged as f16 in LDS (WMMA A layout), A fragments preloaded in VGPRs.
// Per 16-col tile: async global->LDS copy of pre-converted f16 B tile,
// 4x v_wmma_f32_16x16x32_f16, then fused message scatter (fp32 atomics).
// ---------------------------------------------------------------------------
template <int WOUT, int S, bool HASV, bool VEC_OUT, int W2OFF, int W3OFF, int W4OFF>
__global__ __launch_bounds__(256) void edge_conv_kernel(
    const float* __restrict__ emb, const float* __restrict__ es4,
    const float* __restrict__ ev3, const int* __restrict__ ei,
    const float* __restrict__ xs1, const float* __restrict__ xv1,
    const float* __restrict__ R1, const _Float16* __restrict__ B16,
    float* __restrict__ ys, float* __restrict__ yv, int E) {
  constexpr int KP = 132;         // padded K row stride (halves)
  constexpr int WT = WOUT / 16;
  constexpr int TILE_BYTES = 16 * KP * 2;       // 4224
  constexpr int TILE_CHUNKS = TILE_BYTES / 16;  // 264 x b128
  __shared__ float    s_R1[1000];
  __shared__ _Float16 s_A[8][16][KP];                  // per-wave A staging
  __shared__ __align__(16) _Float16 s_Bt[16 * KP];     // shared B tile [n][k]
  __shared__ float    s_w[8][16][16];                  // per-wave w tile
  __shared__ float    s_ee[8][16][8];                  // es[0..3], ev[0..2]

  const int tid = threadIdx.x;
  const int w = tid >> 5;
  const int l = tid & 31;
  const int e0 = (blockIdx.x * 8 + w) * 16;
  const int e_l = l >> 1;
  const int par = l & 1;

  for (int i = tid; i < 1000; i += 256) s_R1[i] = R1[i];
  const int src_r = ei[e0 + e_l];
  const int dst_r = ei[E + e0 + e_l];

  // stage per-edge small operands
  for (int idx = l; idx < 16 * 8; idx += 32) {
    int i = idx >> 3, j = idx & 7;
    float v = 0.f;
    int e = e0 + i;
    if (j < 4) v = es4[(size_t)e * 4 + j];
    else if (j < 7) v = ev3[(size_t)e * 3 + (j - 4)];
    s_ee[w][i][j] = v;
  }
  __syncthreads();  // s_R1 ready

  // hidden layer h = silu(emb @ R1 * rs10) -> f16 A staging
  float em[10];
  {
    const float* er = emb + (size_t)(e0 + e_l) * 10;
#pragma unroll
    for (int k = 0; k < 10; k++) em[k] = er[k];
  }
  for (int jj = 0; jj < 50; jj++) {
    int j = jj * 2 + par;
    float acc = 0.f;
#pragma unroll
    for (int k = 0; k < 10; k++) acc += em[k] * s_R1[k * 100 + j];
    acc *= RS10;
    s_A[w][e_l][j] = (_Float16)siluf_(acc);
  }
  for (int idx = l; idx < 16 * 28; idx += 32) {
    int i = idx / 28, j = 100 + idx % 28;
    s_A[w][i][j] = (_Float16)0.f;
  }

  // preload A fragments for all 4 K-slices (f16 16x32 A layout)
  v16h afr[4];
  {
    const _Float16* arow = &s_A[w][l & 15][0];
    const int abase = ((l >> 4) & 1) * 8;
#pragma unroll
    for (int kk = 0; kk < 4; kk++) {
#pragma unroll
      for (int r = 0; r < 8; r++) {
        int k = kk * 32 + (r >> 2) * 16 + abase + 2 * (r & 3);
        ((unsigned int*)&afr[kk])[r] = *(const unsigned int*)(arow + k);
      }
    }
  }
  const int bbase = ((l >> 4) & 1) * 16;
  const unsigned s_Bt_base = (unsigned)(uintptr_t)(void*)&s_Bt[0];

  for (int t = 0; t < WT; t++) {
    __syncthreads();  // previous tile fully consumed
    // async-copy pre-converted f16 B tile (contiguous 4224B) into LDS
    {
      const void* gbase = (const void*)(B16 + (size_t)t * (16 * KP));
      for (int i = tid; i < TILE_CHUNKS; i += 256)
        async_b128(s_Bt_base + (unsigned)(i * 16), gbase, i * 16);
      wait_async0();
    }
    __syncthreads();  // B tile visible to all waves

    v8f c = {0.f, 0.f, 0.f, 0.f, 0.f, 0.f, 0.f, 0.f};
    {
      const _Float16* brow = &s_Bt[(l & 15) * KP];
#pragma unroll
      for (int kk = 0; kk < 4; kk++) {
        v16h b;
#pragma unroll
        for (int r = 0; r < 8; r++) {
          int k = kk * 32 + (r >> 2) * 16 + (bbase >> 1) + 2 * (r & 3);
          ((unsigned int*)&b)[r] = *(const unsigned int*)(brow + k);
        }
        c = __builtin_amdgcn_wmma_f32_16x16x32_f16(false, afr[kk], false, b,
                                                   (short)0, c, false, false);
      }
    }
    {
      const int n = l & 15, mb = ((l >> 4) & 1) * 8;
#pragma unroll
      for (int r = 0; r < 8; r++) s_w[w][mb + r][n] = c[r];
    }

    // fused message pass for this column tile (2 lanes per edge)
    const int colbase = t * 16;
    if (colbase < 4 * S) {  // w1: mid_s scalar part
      const int ub = colbase >> 2;
#pragma unroll
      for (int q = 0; q < 2; q++) {
        int uu = par * 2 + q;
        int u = ub + uu;
        float sum = 0.f;
#pragma unroll
        for (int vv = 0; vv < 4; vv++) sum += s_w[w][e_l][uu * 4 + vv] * s_ee[w][e_l][vv];
        float xe = xs1[(size_t)src_r * S + u];
        atomAddF(&ys[(size_t)dst_r * 48 + u], 0.5f * xe * sum);
      }
    } else if (VEC_OUT && colbase >= W2OFF && colbase < W2OFF + S) {  // w2: mv1
#pragma unroll
      for (int q = 0; q < 8; q++) {
        int uloc = par * 8 + q;
        int u = (colbase - W2OFF) + uloc;
        float bsv = xs1[(size_t)src_r * S + u] * s_w[w][e_l][uloc];
#pragma unroll
        for (int cc = 0; cc < 3; cc++)
          atomAddF(&yv[(size_t)dst_r * 144 + u * 3 + cc], bsv * s_ee[w][e_l][4 + cc]);
      }
    } else if (HASV && VEC_OUT && colbase >= W3OFF && colbase < W3OFF + 64) {  // w3: mv2
      const int u2b = (colbase - W3OFF) >> 2;
#pragma unroll
      for (int q = 0; q < 2; q++) {
        int uu = par * 2 + q;
        int u2 = u2b + uu;
        float sum = 0.f;
#pragma unroll
        for (int vv = 0; vv < 4; vv++) sum += s_w[w][e_l][uu * 4 + vv] * s_ee[w][e_l][vv];
#pragma unroll
        for (int cc = 0; cc < 3; cc++) {
          float xvv = xv1[(size_t)src_r * 48 + u2 * 3 + cc];
          atomAddF(&yv[(size_t)dst_r * 144 + (S + u2) * 3 + cc], 0.5f * xvv * sum);
        }
      }
    } else if (HASV && colbase >= W4OFF && colbase < W4OFF + 16) {  // w4: mid_s vec part
#pragma unroll
      for (int q = 0; q < 8; q++) {
        int u2 = par * 8 + q;
        float dot = 0.f;
#pragma unroll
        for (int cc = 0; cc < 3; cc++)
          dot += xv1[(size_t)src_r * 48 + u2 * 3 + cc] * s_ee[w][e_l][4 + cc];
        atomAddF(&ys[(size_t)dst_r * 48 + S + u2],
                 s_w[w][e_l][u2] * dot * INV_SQRT3_CONST);
      }
    }
  }
}

// ---------------------------------------------------------------------------
// Node post-pass (layers 1-3): out_s/out_v + gate
// ---------------------------------------------------------------------------
template <int SY, int MV, bool HAS_SCV>
__global__ __launch_bounds__(128) void node_post_kernel(
    const float* __restrict__ sc_scal, const float* __restrict__ sc_vec,
    const float* __restrict__ ys, const float* __restrict__ yv,
    const float* __restrict__ l2s, const float* __restrict__ l2v,
    float* __restrict__ xs_out, float* __restrict__ xv_out, int N) {
  __shared__ float s_l2s[SY * 48];
  __shared__ float s_l2v[MV * 16];
  for (int i = threadIdx.x; i < SY * 48; i += 128) s_l2s[i] = l2s[i];
  for (int i = threadIdx.x; i < MV * 16; i += 128) s_l2v[i] = l2v[i];
  __syncthreads();
  int n = blockIdx.x * 128 + threadIdx.x;
  if (n >= N) return;
  const float scale_s = C_X_CONST * ((SY == 16) ? RS16 : RS48) * 0.25f;
  const float scale_v = (HAS_SCV ? C_X_CONST : 1.0f) * ((MV == 16) ? RS16 : RS48) * 0.25f;
  float yr[SY];
#pragma unroll
  for (int i = 0; i < SY; i++) yr[i] = ys[(size_t)n * 48 + i];
  for (int j = 0; j < 32; j++) {
    float acc = 0.f;
#pragma unroll
    for (int i = 0; i < SY; i++) acc += yr[i] * s_l2s[i * 48 + j];
    float o = C_S_CONST * sc_scal[(size_t)n * 48 + j] + scale_s * acc;
    xs_out[(size_t)n * 32 + j] = siluf_(o);
  }
  for (int ww = 0; ww < 16; ww++) {
    int j = 32 + ww;
    float acc = 0.f;
#pragma unroll
    for (int i = 0; i < SY; i++) acc += yr[i] * s_l2s[i * 48 + j];
    float g = sigmoidf_(C_S_CONST * sc_scal[(size_t)n * 48 + j] + scale_s * acc);
#pragma unroll
    for (int cc = 0; cc < 3; cc++) {
      float accv = 0.f;
      for (int vv = 0; vv < MV; vv++)
        accv += yv[(size_t)n * 144 + vv * 3 + cc] * s_l2v[vv * 16 + ww];
      float t = scale_v * accv;
      if (HAS_SCV) t += C_S_CONST * sc_vec[(size_t)n * 48 + ww * 3 + cc];
      xv_out[(size_t)n * 48 + ww * 3 + cc] = g * t;
    }
  }
}

// ---------------------------------------------------------------------------
// Layer-4 post-pass + output head
// ---------------------------------------------------------------------------
__global__ __launch_bounds__(128) void node_post4_kernel(
    const float* __restrict__ sc_scal, const float* __restrict__ ys,
    const float* __restrict__ l2s, const float* __restrict__ Wh,
    const float* __restrict__ bh, float* __restrict__ out, int N) {
  __shared__ float s_l2s[48 * 8];
  __shared__ float s_wh[24];
  __shared__ float s_bh[3];
  for (int i = threadIdx.x; i < 384; i += 128) s_l2s[i] = l2s[i];
  if (threadIdx.x < 24) s_wh[threadIdx.x] = Wh[threadIdx.x];
  if (threadIdx.x < 3) s_bh[threadIdx.x] = bh[threadIdx.x];
  __syncthreads();
  int n = blockIdx.x * 128 + threadIdx.x;
  if (n >= N) return;
  float yr[48];
#pragma unroll
  for (int i = 0; i < 48; i++) yr[i] = ys[(size_t)n * 48 + i];
  float o[8];
#pragma unroll
  for (int j = 0; j < 8; j++) {
    float acc = 0.f;
#pragma unroll
    for (int i = 0; i < 48; i++) acc += yr[i] * s_l2s[i * 8 + j];
    o[j] = C_S_CONST * sc_scal[(size_t)n * 8 + j] + (C_X_CONST * RS48 * 0.25f) * acc;
  }
#pragma unroll
  for (int k = 0; k < 3; k++) {
    float acc = s_bh[k];
#pragma unroll
    for (int j = 0; j < 8; j++) acc += o[j] * s_wh[j * 3 + k];
    out[(size_t)n * 3 + k] = acc;
  }
}

// ---------------------------------------------------------------------------
extern "C" void kernel_launch(void* const* d_in, const int* in_sizes, int n_in,
                              void* d_out, int out_size, void* d_ws, size_t ws_size,
                              hipStream_t stream) {
  const float* pos       = (const float*)d_in[0];
  const float* onehot    = (const float*)d_in[1];
  const float* eattr     = (const float*)d_in[2];
  const float* W_embed   = (const float*)d_in[3];
  const float* b_embed   = (const float*)d_in[4];
  const float* l1_lin1_s = (const float*)d_in[5];
  const float* l1_sc_s   = (const float*)d_in[6];
  const float* l1_R1     = (const float*)d_in[7];
  const float* l1_R2     = (const float*)d_in[8];
  const float* l1_lin2_s = (const float*)d_in[9];
  const float* l1_lin2_v = (const float*)d_in[10];
  const float* l2_lin1_s = (const float*)d_in[11];
  const float* l2_lin1_v = (const float*)d_in[12];
  const float* l2_sc_s   = (const float*)d_in[13];
  const float* l2_sc_v   = (const float*)d_in[14];
  const float* l2_R1     = (const float*)d_in[15];
  const float* l2_R2     = (const float*)d_in[16];
  const float* l2_lin2_s = (const float*)d_in[17];
  const float* l2_lin2_v = (const float*)d_in[18];
  const float* l3_lin1_s = (const float*)d_in[19];
  const float* l3_lin1_v = (const float*)d_in[20];
  const float* l3_sc_s   = (const float*)d_in[21];
  const float* l3_sc_v   = (const float*)d_in[22];
  const float* l3_R1     = (const float*)d_in[23];
  const float* l3_R2     = (const float*)d_in[24];
  const float* l3_lin2_s = (const float*)d_in[25];
  const float* l3_lin2_v = (const float*)d_in[26];
  const float* l4_lin1_s = (const float*)d_in[27];
  const float* l4_lin1_v = (const float*)d_in[28];
  const float* l4_sc_s   = (const float*)d_in[29];
  const float* l4_R1     = (const float*)d_in[30];
  const float* l4_R2     = (const float*)d_in[31];
  const float* l4_lin2_s = (const float*)d_in[32];
  const float* W_head    = (const float*)d_in[33];
  const float* b_head    = (const float*)d_in[34];
  const int*   ei        = (const int*)d_in[35];
  const int N = in_sizes[0] / 3;
  const int E = in_sizes[2] / 3;  // 800000, multiple of 128
  (void)n_in; (void)out_size; (void)ws_size;

  float* p = (float*)d_ws;
  size_t off = 0;
  auto alloc = [&](size_t cnt) { float* r = p + off; off += (cnt + 3) & ~(size_t)3; return r; };
  float* es   = alloc((size_t)E * 4);
  float* ev   = alloc((size_t)E * 3);
  float* embq = alloc((size_t)E * 10);
  float* xsA  = alloc((size_t)N * 48);
  float* xsB  = alloc((size_t)N * 48);
  float* xvA  = alloc((size_t)N * 48);
  float* xvB  = alloc((size_t)N * 48);
  float* xs1  = alloc((size_t)N * 32);
  float* xv1  = alloc((size_t)N * 48);
  float* scs  = alloc((size_t)N * 48);
  float* scv  = alloc((size_t)N * 48);
  float* ys   = alloc((size_t)N * 48);
  float* yv   = alloc((size_t)N * 144);
  _Float16* b16 = (_Float16*)alloc((size_t)(240 * 132 + 1) / 2);  // f16 B buffer

  dim3 b128(128), b256(256);
  const int nbN = (N + 127) / 128;
  const int nbE128 = E / 128;

  edge_geom_kernel<<<(E + 255) / 256, b256, 0, stream>>>(pos, eattr, ei, es, ev, embq, E);
  embed_kernel<<<(N * 16 + 255) / 256, b256, 0, stream>>>(onehot, W_embed, b_embed, xsA, N);

  // Layer 1 (S=16, no vector input, vec_out)
  node_pre_kernel<16, 48, false, false><<<nbN, b128, 0, stream>>>(
      xsA, nullptr, l1_sc_s, nullptr, l1_lin1_s, nullptr,
      scs, nullptr, xs1, nullptr, ys, yv, N);
  convert_R2_kernel<<<(80 * 132 + 255) / 256, b256, 0, stream>>>(l1_R2, b16, 80);
  edge_conv_kernel<80, 16, false, true, 64, (1 << 28), (1 << 28)>
      <<<nbE128, b256, 0, stream>>>(embq, es, ev, ei, xs1, nullptr,
                                    l1_R1, b16, ys, yv, E);
  node_post_kernel<16, 16, false><<<nbN, b128, 0, stream>>>(
      scs, nullptr, ys, yv, l1_lin2_s, l1_lin2_v, xsB, xvB, N);

  // Layer 2
  node_pre_kernel<32, 48, true, true><<<nbN, b128, 0, stream>>>(
      xsB, xvB, l2_sc_s, l2_sc_v, l2_lin1_s, l2_lin1_v,
      scs, scv, xs1, xv1, ys, yv, N);
  convert_R2_kernel<<<(240 * 132 + 255) / 256, b256, 0, stream>>>(l2_R2, b16, 240);
  edge_conv_kernel<240, 32, true, true, 128, 160, 224>
      <<<nbE128, b256, 0, stream>>>(embq, es, ev, ei, xs1, xv1,
                                    l2_R1, b16, ys, yv, E);
  node_post_kernel<48, 48, true><<<nbN, b128, 0, stream>>>(
      scs, scv, ys, yv, l2_lin2_s, l2_lin2_v, xsA, xvA, N);

  // Layer 3
  node_pre_kernel<32, 48, true, true><<<nbN, b128, 0, stream>>>(
      xsA, xvA, l3_sc_s, l3_sc_v, l3_lin1_s, l3_lin1_v,
      scs, scv, xs1, xv1, ys, yv, N);
  convert_R2_kernel<<<(240 * 132 + 255) / 256, b256, 0, stream>>>(l3_R2, b16, 240);
  edge_conv_kernel<240, 32, true, true, 128, 160, 224>
      <<<nbE128, b256, 0, stream>>>(embq, es, ev, ei, xs1, xv1,
                                    l3_R1, b16, ys, yv, E);
  node_post_kernel<48, 48, true><<<nbN, b128, 0, stream>>>(
      scs, scv, ys, yv, l3_lin2_s, l3_lin2_v, xsB, xvB, N);

  // Layer 4 (no vec_out) + head
  node_pre_kernel<32, 8, true, false><<<nbN, b128, 0, stream>>>(
      xsB, xvB, l4_sc_s, nullptr, l4_lin1_s, l4_lin1_v,
      scs, nullptr, xs1, xv1, ys, yv, N);
  convert_R2_kernel<<<(144 * 132 + 255) / 256, b256, 0, stream>>>(l4_R2, b16, 144);
  edge_conv_kernel<144, 32, true, false, (1 << 28), (1 << 28), 128>
      <<<nbE128, b256, 0, stream>>>(embq, es, ev, ei, xs1, xv1,
                                    l4_R1, b16, ys, yv, E);
  node_post4_kernel<<<nbN, b128, 0, stream>>>(
      scs, ys, l4_lin2_s, W_head, b_head, (float*)d_out, N);
}
